// snn_2_6828998001150
// MI455X (gfx1250) — matrix operand order repaired and checked by reading, not compiled, not run on hardware
//
#include <hip/hip_runtime.h>

typedef _Float16 h16;
typedef __attribute__((ext_vector_type(16))) _Float16 v16h;
typedef __attribute__((ext_vector_type(8)))  _Float16 v8h;
typedef __attribute__((ext_vector_type(8)))  float    v8f;
typedef __attribute__((ext_vector_type(4)))  int      i32x4;

// Problem constants (from reference)
constexpr int T_STEPS = 64;
constexpr int BATCH   = 256;
constexpr int IN_DIM  = 3072;      // C*H*W = 3*32*32
constexpr int H1 = 2048, H2 = 1024, H3 = 512;
constexpr int NCLS = 18;
constexpr int NO_PAD = 32;         // output layer N padded to 32
constexpr int MROWS = T_STEPS * BATCH;  // 16384 GEMM rows

constexpr size_t MiB = 1024ull * 1024ull;

// CDNA5 async global->LDS staging (ASYNCcnt-tracked), if the toolchain
// exposes the gfx1250 builtins. Falls back to sync VGPR-bounce staging.
#if defined(__gfx1250__) && \
    __has_builtin(__builtin_amdgcn_global_load_async_to_lds_b128) && \
    __has_builtin(__builtin_amdgcn_s_wait_asynccnt)
#define USE_ASYNC_LDS 1
#define ASYNC_GPTR(p) ((__attribute__((address_space(1))) i32x4*)(p))
#define ASYNC_LPTR(p) ((__attribute__((address_space(3))) i32x4*)(p))
#else
#define USE_ASYNC_LDS 0
#endif

// ---------------------------------------------------------------------------
// Tiled WMMA GEMM:  C[M,N] = A[M,K] @ W[N,K]^T + bias[N]
// A, W f16 row-major; C written f16; f32 accumulation via
// V_WMMA_F32_16X16X32_F16 (wave32).
// Block = 256 threads = 8 waves. Block tile BM x BN, K step 64,
// double-buffered LDS staged with async copies (prefetch next K block
// while WMMAs consume the current one).
// ---------------------------------------------------------------------------
template<int BM, int BN, int WROWS, int WCOLS>
__global__ __launch_bounds__(256)
void gemm_wmma_f16(const h16* __restrict__ A, const h16* __restrict__ W,
                   const float* __restrict__ bias, h16* __restrict__ C,
                   int M, int N, int K)
{
    constexpr int KS = 64;           // K step per stage
    constexpr int WM = BM / WROWS;   // rows per wave
    constexpr int WN = BN / WCOLS;   // cols per wave
    constexpr int TM = WM / 16;      // 16x16 tiles per wave (rows)
    constexpr int TN = WN / 16;      // 16x16 tiles per wave (cols)
    constexpr int ACH = BM * (KS / 8);          // 8-half chunks in A block
    constexpr int BCH = BN * (KS / 8);          // 8-half chunks in B block
    constexpr int AIT = ACH / 256;              // chunks per thread (A)
    constexpr int BIT = BCH / 256;              // chunks per thread (B), >=1
    constexpr int NPREF = AIT + BIT;            // async ops per stage per wave
    static_assert(ACH % 256 == 0 && BCH % 256 == 0, "chunk counts");

    __shared__ __align__(32) h16 sA[2][BM * KS];
    __shared__ __align__(32) h16 sB[2][BN * KS];

    const int tid  = threadIdx.x;
    const int lane = tid & 31;
    const int wave = tid >> 5;
    const int wr   = wave / WCOLS;
    const int wc   = wave % WCOLS;
    const int l16  = lane & 15;
    const int hsel = lane >> 4;      // 0: lanes 0-15, 1: lanes 16-31

    const int m0 = blockIdx.y * BM;
    const int n0 = blockIdx.x * BN;

    // Stage one BM/BN x KS block into LDS buffer `buf` (async if available).
    auto stage = [&](int k0, int buf) {
        #pragma unroll
        for (int i = 0; i < AIT; i++) {
            int c = tid + i * 256;
            int m = c >> 3, kk = (c & 7) * 8;
            const h16* g = &A[(size_t)(m0 + m) * K + k0 + kk];
            h16* l = &sA[buf][m * KS + kk];
#if USE_ASYNC_LDS
            __builtin_amdgcn_global_load_async_to_lds_b128(
                ASYNC_GPTR(g), ASYNC_LPTR(l), 0, 0);
#else
            *(v8h*)l = *(const v8h*)g;
#endif
        }
        #pragma unroll
        for (int i = 0; i < BIT; i++) {
            int c = tid + i * 256;
            int n = c >> 3, kk = (c & 7) * 8;
            const h16* g = &W[(size_t)(n0 + n) * K + k0 + kk];
            h16* l = &sB[buf][n * KS + kk];
#if USE_ASYNC_LDS
            __builtin_amdgcn_global_load_async_to_lds_b128(
                ASYNC_GPTR(g), ASYNC_LPTR(l), 0, 0);
#else
            *(v8h*)l = *(const v8h*)g;
#endif
        }
    };

    v8f acc[TM][TN];
    #pragma unroll
    for (int ti = 0; ti < TM; ti++)
        #pragma unroll
        for (int tj = 0; tj < TN; tj++)
            #pragma unroll
            for (int e = 0; e < 8; e++)
                acc[ti][tj][e] = 0.0f;

    stage(0, 0);

    for (int k0 = 0; k0 < K; k0 += KS) {
        const int buf = (k0 / KS) & 1;
        if (k0 + KS < K) {
            stage(k0 + KS, buf ^ 1);      // prefetch next block
#if USE_ASYNC_LDS
            __builtin_amdgcn_s_wait_asynccnt(NPREF);  // current block landed
#endif
        } else {
#if USE_ASYNC_LDS
            __builtin_amdgcn_s_wait_asynccnt(0);
#endif
        }
        __syncthreads();

        // Two 16x16x32 sub-steps over the KS=64 block.
        #pragma unroll
        for (int sub = 0; sub < 2; sub++) {
            const int kb = sub * 32;
            // A fragments: lane row M=l16; lanes 0-15 K=0..15,
            // lanes 16-31 K=16..31  (same k order as B)
            v16h afr[TM];
            #pragma unroll
            for (int ti = 0; ti < TM; ti++) {
                int r = wr * WM + ti * 16 + l16;
                afr[ti] = *(const v16h*)&sA[buf][r * KS + kb + hsel * 16];
            }
            // B fragments: lane col N=l16; lanes 0-15 K=0..15, lanes 16-31
            // K=16..31 (contiguous in W row-major since B = W^T)
            v16h bfr[TN];
            #pragma unroll
            for (int tj = 0; tj < TN; tj++) {
                int n = wc * WN + tj * 16 + l16;
                bfr[tj] = *(const v16h*)&sB[buf][n * KS + kb + hsel * 16];
            }
            #pragma unroll
            for (int ti = 0; ti < TM; ti++)
                #pragma unroll
                for (int tj = 0; tj < TN; tj++)
                    acc[ti][tj] = __builtin_amdgcn_wmma_f32_16x16x32_f16(
                        false, afr[ti], false, bfr[tj],
                        (short)0, acc[ti][tj], false, false);
        }
        __syncthreads();   // protect buf^1 before next stage overwrites it
    }

    // Epilogue: C/D layout (ISA 7.12.2): vgpr r, lane l -> row r + 8*(l>=16),
    // col l%16. Add bias, convert to f16.
    #pragma unroll
    for (int ti = 0; ti < TM; ti++) {
        #pragma unroll
        for (int tj = 0; tj < TN; tj++) {
            int gr = m0 + wr * WM + ti * 16 + hsel * 8;
            int gc = n0 + wc * WN + tj * 16 + l16;
            float bv = bias[gc];
            #pragma unroll
            for (int r = 0; r < 8; r++)
                C[(size_t)(gr + r) * N + gc] = (h16)(acc[ti][tj][r] + bv);
        }
    }
}

// ---------------------------------------------------------------------------
// LIF recurrence (snntorch Leaky, reset-by-subtraction, detached reset):
//   reset = (mem > thr) ? thr : 0        (pre-update membrane)
//   mem   = beta*mem + cur - reset
//   spk   = (mem > thr) ? 1 : 0
// One thread per (batch, neuron); serial loop over T in registers.
// ---------------------------------------------------------------------------
__global__ __launch_bounds__(256)
void lif_kernel(const h16* __restrict__ cur, const float* __restrict__ beta,
                const float* __restrict__ thr, h16* __restrict__ spk,
                int BN, int Nh)
{
    int idx = blockIdx.x * blockDim.x + threadIdx.x;
    if (idx >= BN) return;
    int n = idx - (idx / Nh) * Nh;
    float be = beta[n], th = thr[n];
    float mem = 0.0f;
    for (int t = 0; t < T_STEPS; t++) {
        float c = (float)cur[(size_t)t * BN + idx];
        float reset = (mem > th) ? th : 0.0f;
        mem = be * mem + c - reset;
        spk[(size_t)t * BN + idx] = (mem > th) ? (h16)1.0f : (h16)0.0f;
    }
}

// Output LIF: scalar threshold, 18 classes, cur padded to stride 32,
// final spikes written as f32 to d_out [T, B, 18].
__global__ __launch_bounds__(256)
void lif_out_kernel(const h16* __restrict__ cur, const float* __restrict__ beta,
                    const float* __restrict__ thr_o, float* __restrict__ out)
{
    int idx = blockIdx.x * blockDim.x + threadIdx.x;
    if (idx >= BATCH * NCLS) return;
    int b = idx / NCLS, n = idx - b * NCLS;
    float be = beta[n], th = thr_o[0];
    float mem = 0.0f;
    for (int t = 0; t < T_STEPS; t++) {
        float c = (float)cur[((size_t)t * BATCH + b) * NO_PAD + n];
        float reset = (mem > th) ? th : 0.0f;
        mem = be * mem + c - reset;
        out[((size_t)t * BATCH + b) * NCLS + n] = (mem > th) ? 1.0f : 0.0f;
    }
}

// ---------------------------------------------------------------------------
// x [B,C,T,H,W] f32  ->  X [T, B, C*H*W] f16
// ---------------------------------------------------------------------------
__global__ __launch_bounds__(256)
void x_to_f16(const float* __restrict__ x, h16* __restrict__ xt)
{
    constexpr size_t TOTAL = (size_t)T_STEPS * BATCH * IN_DIM;
    size_t idx = (size_t)blockIdx.x * blockDim.x + threadIdx.x;
    if (idx >= TOTAL) return;
    int f = (int)(idx % IN_DIM);
    size_t tb = idx / IN_DIM;
    int b = (int)(tb % BATCH);
    int t = (int)(tb / BATCH);
    int c  = f / (32 * 32);
    int hw = f - c * (32 * 32);
    size_t src = (((size_t)b * 3 + c) * T_STEPS + t) * (32 * 32) + hw;
    xt[idx] = (h16)x[src];
}

__global__ __launch_bounds__(256)
void f32_to_f16(const float* __restrict__ src, h16* __restrict__ dst, int n)
{
    int i = blockIdx.x * blockDim.x + threadIdx.x;
    if (i < n) dst[i] = (h16)src[i];
}

// Pad Wo [18,512] -> [32,512] f16 (zero rows 18..31) and bo [18] -> [32] f32.
__global__ __launch_bounds__(256)
void pad_wo(const float* __restrict__ Wo, const float* __restrict__ bo,
            h16* __restrict__ Woh, float* __restrict__ boh)
{
    int i = blockIdx.x * blockDim.x + threadIdx.x;
    if (i >= NO_PAD * H3) return;
    int r = i / H3, k = i - r * H3;
    Woh[i] = (r < NCLS) ? (h16)Wo[r * H3 + k] : (h16)0.0f;
    if (k == 0) boh[r] = (r < NCLS) ? bo[r] : 0.0f;
}

// ---------------------------------------------------------------------------
// Workspace layout (bytes):
//   [0,12MiB)      W1 f16 (2048x3072)
//   [12,16MiB)     W2 f16 (1024x2048)
//   [16,17MiB)     W3 f16 (512x1024)
//   [17MiB,+32KB)  Wo f16 padded (32x512); bias pad at +64KB
//   [32,128MiB)    R1: X f16 (96MiB) -> CUR2 -> S3
//   [128,192MiB)   R2: CUR1 (64MiB) -> S2 -> CURo
//   [192,256MiB)   R3: S1 (64MiB) -> CUR3
// Peak: 256 MiB.
// ---------------------------------------------------------------------------
extern "C" void kernel_launch(void* const* d_in, const int* in_sizes, int n_in,
                              void* d_out, int out_size, void* d_ws, size_t ws_size,
                              hipStream_t stream)
{
    const float* x      = (const float*)d_in[0];
    const float* W1     = (const float*)d_in[1];
    const float* b1     = (const float*)d_in[2];
    const float* beta1  = (const float*)d_in[3];
    const float* thr1   = (const float*)d_in[4];
    const float* W2     = (const float*)d_in[5];
    const float* b2     = (const float*)d_in[6];
    const float* beta2  = (const float*)d_in[7];
    const float* thr2   = (const float*)d_in[8];
    const float* W3     = (const float*)d_in[9];
    const float* b3     = (const float*)d_in[10];
    const float* beta3  = (const float*)d_in[11];
    const float* thr3   = (const float*)d_in[12];
    const float* Wo     = (const float*)d_in[13];
    const float* bo     = (const float*)d_in[14];
    const float* beta_o = (const float*)d_in[15];
    const float* thr_o  = (const float*)d_in[16];
    float* out = (float*)d_out;

    char* ws = (char*)d_ws;
    h16*   W1h = (h16*)(ws + 0);
    h16*   W2h = (h16*)(ws + 12 * MiB);
    h16*   W3h = (h16*)(ws + 16 * MiB);
    h16*   Woh = (h16*)(ws + 17 * MiB);
    float* boh = (float*)(ws + 17 * MiB + 64 * 1024);
    h16*   R1  = (h16*)(ws + 32 * MiB);
    h16*   R2  = (h16*)(ws + 128 * MiB);
    h16*   R3  = (h16*)(ws + 192 * MiB);

    // --- weight / input precision conversion ---
    f32_to_f16<<<(H1 * IN_DIM + 255) / 256, 256, 0, stream>>>(W1, W1h, H1 * IN_DIM);
    f32_to_f16<<<(H2 * H1 + 255) / 256, 256, 0, stream>>>(W2, W2h, H2 * H1);
    f32_to_f16<<<(H3 * H2 + 255) / 256, 256, 0, stream>>>(W3, W3h, H3 * H2);
    pad_wo<<<(NO_PAD * H3 + 255) / 256, 256, 0, stream>>>(Wo, bo, Woh, boh);
    {
        size_t totalX = (size_t)T_STEPS * BATCH * IN_DIM;
        x_to_f16<<<(unsigned)((totalX + 255) / 256), 256, 0, stream>>>(x, R1);
    }

    // --- layer 1: CUR1 = X @ W1^T + b1 ; LIF1 -> S1 ---
    gemm_wmma_f16<128, 128, 4, 2>
        <<<dim3(H1 / 128, MROWS / 128), 256, 0, stream>>>(
            R1, W1h, b1, R2, MROWS, H1, IN_DIM);
    lif_kernel<<<(BATCH * H1) / 256, 256, 0, stream>>>(
        R2, beta1, thr1, R3, BATCH * H1, H1);

    // --- layer 2 ---
    gemm_wmma_f16<128, 128, 4, 2>
        <<<dim3(H2 / 128, MROWS / 128), 256, 0, stream>>>(
            R3, W2h, b2, R1, MROWS, H2, H1);
    lif_kernel<<<(BATCH * H2) / 256, 256, 0, stream>>>(
        R1, beta2, thr2, R2, BATCH * H2, H2);

    // --- layer 3 ---
    gemm_wmma_f16<128, 128, 4, 2>
        <<<dim3(H3 / 128, MROWS / 128), 256, 0, stream>>>(
            R2, W3h, b3, R3, MROWS, H3, H2);
    lif_kernel<<<(BATCH * H3) / 256, 256, 0, stream>>>(
        R3, beta3, thr3, R1, BATCH * H3, H3);

    // --- output layer (N padded to 32) ---
    gemm_wmma_f16<128, 32, 8, 1>
        <<<dim3(1, MROWS / 128), 256, 0, stream>>>(
            R1, Woh, boh, R2, MROWS, NO_PAD, H3);
    lif_out_kernel<<<(BATCH * NCLS + 255) / 256, 256, 0, stream>>>(
        R2, beta_o, thr_o, out);
}
